// SSDBoxHead_86861418594860
// MI455X (gfx1250) — compile-verified, matrix operand order checked by reading, not run on hardware
//
#include <hip/hip_runtime.h>
#include <math.h>
#include <stdint.h>

// ---------------- problem constants (match reference) ----------------
#define B_IMG   32
#define N_PRI   8732
#define N_CLS   81
#define CM1     80
#define S_FLAT  (N_PRI * CM1)      // 698560 scores per image (background dropped)
#define TOPK    1000
#define CAND    1024               // padded candidate capacity (power of two for bitonic)
#define MAXOUT  100
#define CONF_THR 0.01f
#define NMS_THR  0.45f
#define IMG_SZ   300.0f
#define CLS_OFF  (2.0f * IMG_SZ)   // per-class coordinate offset for batched NMS
#define NBIN     1024

typedef __attribute__((ext_vector_type(16))) _Float16 v16h;
typedef __attribute__((ext_vector_type(8)))  float    v8f;

#define WROW   1312   // per-wave LDS floats: max(16*81 stage = 1296, 16*80 scores = 1280), 16B aligned

// =====================================================================
// Kernel 1: fused softmax (WMMA row-sum) + confidence threshold + decode
// One wave32 handles 16 rows x 81 classes.
//  - logits staged global->LDS with GLOBAL_LOAD_ASYNC_TO_LDS_B128 (ASYNCcnt)
//  - row sums via D = A x ones: 3x v_wmma_f32_16x16x32_f16 per wave
//  - thresholded scores staged in LDS, written back as coalesced b128
// =====================================================================
__global__ __launch_bounds__(256)
void k_softmax_decode(const float* __restrict__ logits,
                      const float* __restrict__ bbox,
                      const float* __restrict__ priors,
                      float* __restrict__ scoresOut,   // flat [B*N][80], -1 if below thr
                      float* __restrict__ boxesOut)    // [B*N][4] xyxy pixels
{
    __shared__ float lds[8 * WROW];                    // 41 KB
    const int wid  = threadIdx.x >> 5;
    const int lane = threadIdx.x & 31;
    const int baseRow = (blockIdx.x * 8 + wid) * 16;   // grid sized exactly
    if (baseRow >= B_IMG * N_PRI) return;              // uniform per wave (never fires)
    float* wl = lds + wid * WROW;

    // ---- async stage: 16 rows x 81 logits = 324 float4 -> LDS (contiguous) ----
    const float* src = logits + (size_t)baseRow * N_CLS;   // 16B aligned (baseRow*324B)
    {
        // wave-uniform 64-bit scalar base for GVS addressing
        uint64_t sa = (uint64_t)(uintptr_t)src;
        uint32_t salo = __builtin_amdgcn_readfirstlane((uint32_t)sa);
        uint32_t sahi = __builtin_amdgcn_readfirstlane((uint32_t)(sa >> 32));
        uint64_t sbase = ((uint64_t)sahi << 32) | salo;
        uint32_t ldsBase = (uint32_t)(uintptr_t)wl;        // low 32 bits = LDS byte offset
#pragma unroll
        for (int it = 0; it < 10; ++it) {                  // 10*32 = 320 float4
            uint32_t vOff = (uint32_t)((it * 32 + lane) * 16);
            uint32_t ldsA = ldsBase + vOff;
            asm volatile("global_load_async_to_lds_b128 %0, %1, %2"
                         :: "v"(ldsA), "v"(vOff), "s"(sbase) : "memory");
        }
        if (lane < 4) {                                    // tail: 4 float4
            float4 t = ((const float4*)src)[320 + lane];
            ((float4*)wl)[320 + lane] = t;
        }
        asm volatile("s_wait_asynccnt 0x0" ::: "memory");
    }
    __syncthreads();

    // ISA 16-bit A layout: lane<16 owns K={j, j+8}(j<8 / j>=8), lane>=16 owns +8
    const int r  = lane & 15;
    const int hi = (lane >= 16) ? 8 : 0;

    // row max (this lane's 48 slots, then combine halves)
    float rowmax = -INFINITY;
#pragma unroll
    for (int kc = 0; kc < 3; ++kc)
#pragma unroll
        for (int j = 0; j < 16; ++j) {
            int col = kc * 32 + j + ((j < 8) ? 0 : 8) + hi;
            float v = (col < N_CLS) ? wl[r * N_CLS + col] : -INFINITY;
            rowmax = fmaxf(rowmax, v);
        }
    rowmax = fmaxf(rowmax, __shfl_xor(rowmax, 16));

    v16h ones;
#pragma unroll
    for (int j = 0; j < 16; ++j) ones[j] = (_Float16)1.0f;

    v8f  acc = {};
    v16h aChunk[3];
#pragma unroll
    for (int kc = 0; kc < 3; ++kc) {
        v16h a;
#pragma unroll
        for (int j = 0; j < 16; ++j) {
            int col = kc * 32 + j + ((j < 8) ? 0 : 8) + hi;
            float v = (col < N_CLS) ? wl[r * N_CLS + col] : -INFINITY;
            a[j] = (_Float16)__expf(v - rowmax);
        }
        aChunk[kc] = a;
        // D += A x ones  ==> every D[m][n] = row-sum of A row m
        acc = __builtin_amdgcn_wmma_f32_16x16x32_f16(false, a, false, ones,
                                                     (short)0, acc, false, false);
    }

    // D layout: lanes 0-15 hold M=0..7 in acc[0..7]; lanes 16-31 hold M=8..15.
    const int ri = r & 7;
    float candv =
        (ri == 0) ? acc[0] : (ri == 1) ? acc[1] : (ri == 2) ? acc[2] :
        (ri == 3) ? acc[3] : (ri == 4) ? acc[4] : (ri == 5) ? acc[5] :
        (ri == 6) ? acc[6] : acc[7];
    float other  = __shfl_xor(candv, 16);
    float rowsum = ((r < 8) == (lane < 16)) ? candv : other;
    float invsum = 1.0f / rowsum;

    // ---- stage thresholded scores into LDS (reuse wl): [r][col-1], 16x80 ----
    __syncthreads();
#pragma unroll
    for (int kc = 0; kc < 3; ++kc)
#pragma unroll
        for (int j = 0; j < 16; ++j) {
            int col = kc * 32 + j + ((j < 8) ? 0 : 8) + hi;
            if (col >= 1 && col < N_CLS) {
                float p = (float)aChunk[kc][j] * invsum;
                wl[r * CM1 + (col - 1)] = (p > CONF_THR) ? p : -1.0f;
            }
        }
    __syncthreads();

    // ---- coalesced write-back: scoresOut is flat [B*N][80] => contiguous span ----
    {
        float4* dst4 = (float4*)(scoresOut + (size_t)baseRow * CM1);
        const float4* wl4 = (const float4*)wl;
        for (int i = lane; i < (16 * CM1) / 4; i += 32)    // 320 float4
            dst4[i] = wl4[i];
    }

    // ---- decode boxes: lanes 0..15 handle one row each ----
    if (lane < 16) {
        int g2 = baseRow + lane;
        int b2 = g2 / N_PRI;
        int n2 = g2 - b2 * N_PRI;
        const float4 t  = ((const float4*)bbox)[g2];
        const float4 pr = ((const float4*)priors)[n2];
        float cx = t.x * 0.1f * pr.z + pr.x;
        float cy = t.y * 0.1f * pr.w + pr.y;
        float w  = __expf(t.z * 0.2f) * pr.z;
        float h  = __expf(t.w * 0.2f) * pr.w;
        float4 o;
        o.x = (cx - 0.5f * w) * IMG_SZ;
        o.y = (cy - 0.5f * h) * IMG_SZ;
        o.z = (cx + 0.5f * w) * IMG_SZ;
        o.w = (cy + 0.5f * h) * IMG_SZ;
        ((float4*)boxesOut)[g2] = o;
    }
}

// =====================================================================
// Kernel 2: per-image histogram select -> deterministic compaction ->
// bitonic sort -> greedy class-aware NMS -> top-100 output.
// One 1024-thread block per image; score row (2.8 MB) is L2-resident.
// =====================================================================
__global__ __launch_bounds__(1024)
void k_select_nms(const float* __restrict__ scoresBuf,
                  const float* __restrict__ boxesBuf,
                  float* __restrict__ out)
{
    __shared__ int   auxA[CAND];                 // histogram -> suppression flags
    __shared__ int   auxB[CAND];                 // keep flags
    __shared__ float aS[CAND];  __shared__ int aI[CAND];   // strictly-above / cand list
    __shared__ float eS[CAND];  __shared__ int eI[CAND];   // cutoff-bin list
    __shared__ float bx0[CAND], by0[CAND], bx2[CAND], by2[CAND]; // offset boxes
    __shared__ int   lab[CAND];
    __shared__ int   wsum[33];
    __shared__ int   s_cut, s_nA, s_nE;
    __shared__ float outB[MAXOUT * 4], outS[MAXOUT], outL[MAXOUT];

    const int tid  = threadIdx.x;
    const int lane = tid & 31, wid = tid >> 5;
    const int b    = blockIdx.x;
    const float* simg = scoresBuf + (size_t)b * S_FLAT;

    // ---- phase 1: 1024-bin histogram of kept scores ----
    auxA[tid] = 0;
    __syncthreads();
    for (int t = tid; t < S_FLAT; t += 1024) {
        __builtin_prefetch(&simg[t + 8192], 0, 0);   // global_prefetch_b8 (L2-resident row)
        float s = simg[t];
        if (s > 0.0f) {
            int bin = (int)(s * (float)NBIN);
            bin = bin > (NBIN - 1) ? (NBIN - 1) : bin;
            atomicAdd(&auxA[bin], 1);
        }
    }
    __syncthreads();

    // ---- phase 2: find cutoff bin (suffix count >= TOPK) ----
    if (tid == 0) {
        int acc = 0, cut = 0;
        for (int i = NBIN - 1; i >= 0; --i) {
            acc += auxA[i];
            if (acc >= TOPK) { cut = i; break; }
        }
        s_cut = cut; s_nA = 0; s_nE = 0;
    }
    __syncthreads();
    const int cut = s_cut;

    // ---- phase 3: deterministic (index-ordered) compaction via wave32 ballots ----
    const int nChunk = (S_FLAT + 1023) / 1024;
    for (int ch = 0; ch < nChunk; ++ch) {
        int t = ch * 1024 + tid;
        float s = (t < S_FLAT) ? simg[t] : -1.0f;
        int bin = -1;
        if (s > 0.0f) {
            bin = (int)(s * (float)NBIN);
            bin = bin > (NBIN - 1) ? (NBIN - 1) : bin;
        }
        // strictly above cutoff (guaranteed < TOPK total)
        {
            bool f = (bin > cut);
            unsigned m = (unsigned)__ballot(f);
            int lp = __popc(m & ((1u << lane) - 1u));
            if (lane == 0) wsum[wid] = __popc(m);
            __syncthreads();
            if (tid == 0) {
                int a2 = 0;
                for (int w = 0; w < 32; ++w) { int v = wsum[w]; wsum[w] = a2; a2 += v; }
                wsum[32] = a2;
            }
            __syncthreads();
            int pos = s_nA + wsum[wid] + lp;
            if (f && pos < CAND) { aS[pos] = s; aI[pos] = t; }
            __syncthreads();
            if (tid == 0) s_nA = min(s_nA + wsum[32], CAND);
            __syncthreads();
        }
        // equal to cutoff (capped, index order)
        {
            bool f = (bin == cut) && (bin >= 0);
            unsigned m = (unsigned)__ballot(f);
            int lp = __popc(m & ((1u << lane) - 1u));
            if (lane == 0) wsum[wid] = __popc(m);
            __syncthreads();
            if (tid == 0) {
                int a2 = 0;
                for (int w = 0; w < 32; ++w) { int v = wsum[w]; wsum[w] = a2; a2 += v; }
                wsum[32] = a2;
            }
            __syncthreads();
            int pos = s_nE + wsum[wid] + lp;
            if (f && pos < CAND) { eS[pos] = s; eI[pos] = t; }
            __syncthreads();
            if (tid == 0) s_nE = min(s_nE + wsum[32], CAND);
            __syncthreads();
        }
    }

    // ---- build candidate list (<= TOPK) into aS/aI, padded with -1 ----
    int nA    = min(s_nA, TOPK);
    int nEuse = min(s_nE, TOPK - nA);
    float myS = -1.0f; int myI = 0;
    if (tid < nA)               { myS = aS[tid];      myI = aI[tid]; }
    else if (tid < nA + nEuse)  { myS = eS[tid - nA]; myI = eI[tid - nA]; }
    __syncthreads();
    aS[tid] = myS; aI[tid] = myI;
    __syncthreads();

    // ---- phase 4: bitonic sort, score desc, index asc tiebreak ----
    for (int k = 2; k <= CAND; k <<= 1) {
        for (int js = k >> 1; js > 0; js >>= 1) {
            int ixj = tid ^ js;
            if (ixj > tid) {
                float s0 = aS[tid], s1 = aS[ixj];
                int   i0 = aI[tid], i1 = aI[ixj];
                bool desc = ((tid & k) == 0);
                bool sw = desc ? (s0 < s1 || (s0 == s1 && i0 > i1))
                               : (s0 > s1 || (s0 == s1 && i0 < i1));
                if (sw) { aS[tid] = s1; aS[ixj] = s0; aI[tid] = i1; aI[ixj] = i0; }
            }
            __syncthreads();
        }
    }

    // ---- phase 5: gather candidate boxes with per-class offsets ----
    {
        float sc = aS[tid]; int idx = aI[tid];
        int n = idx / CM1;
        int l = idx - n * CM1 + 1;
        float off = (float)l * CLS_OFF;
        float4 bb = make_float4(0.f, 0.f, 0.f, 0.f);
        if (sc > 0.0f) bb = ((const float4*)boxesBuf)[b * N_PRI + n];
        bx0[tid] = bb.x + off; by0[tid] = bb.y + off;
        bx2[tid] = bb.z + off; by2[tid] = bb.w + off;
        lab[tid] = (sc > 0.0f) ? l : 0;
        auxA[tid] = 0;  // suppression
        auxB[tid] = 0;  // keep
    }
    __syncthreads();

    // ---- phase 6: greedy NMS (sequential outer, parallel suppression) ----
    for (int i = 0; i < TOPK; ++i) {
        bool vi  = (auxA[i] == 0) && (aS[i] > 0.0f);
        float x0 = bx0[i], y0 = by0[i], x1 = bx2[i], y1 = by2[i];
        __syncthreads();                         // all reads of slot i done before writes
        if (tid == 0 && vi) auxB[i] = 1;
        if (vi) {
            float xx0 = fmaxf(x0, bx0[tid]);
            float yy0 = fmaxf(y0, by0[tid]);
            float xx1 = fminf(x1, bx2[tid]);
            float yy1 = fminf(y1, by2[tid]);
            float iw = fmaxf(xx1 - xx0, 0.0f), ih = fmaxf(yy1 - yy0, 0.0f);
            float inter = iw * ih;
            float a1 = (x1 - x0) * (y1 - y0);
            float a2 = (bx2[tid] - bx0[tid]) * (by2[tid] - by0[tid]);
            float iou = inter / (a1 + a2 - inter + 1e-9f);
            if (iou > NMS_THR) auxA[tid] = 1;
        }
        __syncthreads();
    }

    // ---- phase 7: rank kept candidates, emit top-100 ----
    int f = auxB[tid];
    unsigned m = (unsigned)__ballot(f != 0);
    int lp = __popc(m & ((1u << lane) - 1u));
    if (lane == 0) wsum[wid] = __popc(m);
    __syncthreads();
    if (tid == 0) {
        int a2 = 0;
        for (int w = 0; w < 32; ++w) { int v = wsum[w]; wsum[w] = a2; a2 += v; }
    }
    __syncthreads();
    int rank = wsum[wid] + lp;

    if (tid < MAXOUT)     { outS[tid] = 0.0f; outL[tid] = 0.0f; }
    if (tid < MAXOUT * 4) outB[tid] = 0.0f;
    __syncthreads();
    if (f && rank < MAXOUT) {
        float off = (float)lab[tid] * CLS_OFF;
        outB[rank * 4 + 0] = bx0[tid] - off;
        outB[rank * 4 + 1] = by0[tid] - off;
        outB[rank * 4 + 2] = bx2[tid] - off;
        outB[rank * 4 + 3] = by2[tid] - off;
        outS[rank] = aS[tid];
        outL[rank] = (float)lab[tid];
    }
    __syncthreads();

    float* oB = out;                              // [32][100][4]
    float* oS = out + B_IMG * MAXOUT * 4;         // [32][100]
    float* oL = oS + B_IMG * MAXOUT;              // [32][100]
    if (tid < MAXOUT * 4) oB[b * MAXOUT * 4 + tid] = outB[tid];
    if (tid < MAXOUT) {
        oS[b * MAXOUT + tid] = outS[tid];
        oL[b * MAXOUT + tid] = outL[tid];
    }
}

// =====================================================================
extern "C" void kernel_launch(void* const* d_in, const int* in_sizes, int n_in,
                              void* d_out, int out_size, void* d_ws, size_t ws_size,
                              hipStream_t stream)
{
    const float* logits = (const float*)d_in[0];   // [32,8732,81]
    const float* bbox   = (const float*)d_in[1];   // [32,8732,4]
    const float* priors = (const float*)d_in[2];   // [8732,4]

    // workspace: scores flat [B*N][80] (89.4 MB, L2-resident) then boxes [B*N][4]
    float* scoresBuf = (float*)d_ws;
    float* boxesBuf  = scoresBuf + (size_t)B_IMG * S_FLAT;

    const int totalRows = B_IMG * N_PRI;           // 279424, divisible by 16
    const int waves  = totalRows / 16;             // 17464
    const int blocks = (waves + 7) / 8;            // 2183 (exact)

    k_softmax_decode<<<blocks, 256, 0, stream>>>(logits, bbox, priors,
                                                 scoresBuf, boxesBuf);
    k_select_nms<<<B_IMG, 1024, 0, stream>>>(scoresBuf, boxesBuf, (float*)d_out);
}
